// BottleneckLSTMCell_83021717832024
// MI455X (gfx1250) — matrix operand order repaired and checked by reading, not compiled, not run on hardware
//
#include <hip/hip_runtime.h>
#include <stdint.h>

typedef __attribute__((ext_vector_type(16))) _Float16     v16h;
typedef __attribute__((ext_vector_type(8)))  _Float16     v8h;
typedef __attribute__((ext_vector_type(8)))  float        v8f;
typedef __attribute__((ext_vector_type(4)))  unsigned int v4u;
typedef __attribute__((ext_vector_type(8)))  int          v8i;
typedef __attribute__((ext_vector_type(4)))  int          v4i;

#define HWDIM 64
#define NPIX  4096
#define CIN_  320
#define CH_   512
#define KCAT  832          // CIN_ + CH_
#define BATCH 8

#define TN 64              // workgroup N tile (pixels)
#define TK 32              // K step (one WMMA f16 K)
#define TM1 128            // gemm1 M tile (8 waves x 16 rows)
#define TMG 64             // gates M tile (4 waves x 16 rows, 2 N-halves)

#if __has_builtin(__builtin_amdgcn_tensor_load_to_lds) && __has_builtin(__builtin_amdgcn_s_wait_tensorcnt)
#define HAVE_TDM 1
#else
#define HAVE_TDM 0
#endif

// ---------------------------------------------------------------------------
// TDM: build D# per CDNA5 ISA 8.3/8.4 and issue TENSOR_LOAD_TO_LDS (f16 data).
// Assumes the kernel's single __shared__ array sits at group-segment offset 0.
// ---------------------------------------------------------------------------
#if HAVE_TDM
__device__ __forceinline__ void tdm_load_f16(
    uint32_t lds_byte, uint64_t gaddr,
    uint32_t td0, uint32_t td1, uint32_t td2,     // tensor dims (elems)
    uint64_t s0, uint64_t s1,                     // dim0/dim1 strides (elems)
    uint32_t t0, uint32_t t1, uint32_t t2)        // tile dims (elems)
{
  v4u g0;
  g0[0] = 1u;                                        // count=1, user descriptor
  g0[1] = lds_byte;                                  // lds_addr (bytes)
  g0[2] = (uint32_t)gaddr;                           // global_addr[31:0]
  g0[3] = (uint32_t)((gaddr >> 32) & 0x01FFFFFFu)    // global_addr[56:32]
        | (2u << 30);                                // type=2 ("image")
  v8i g1;
  g1[0] = (int)(1u << 16);                           // wg_mask=0 | data_size=1(2B)
  g1[1] = (int)((td0 & 0xFFFFu) << 16);              // abar=0 | tensor_dim0 lo16
  g1[2] = (int)((td0 >> 16) | ((td1 & 0xFFFFu) << 16));
  g1[3] = (int)((td1 >> 16) | (t0 << 16));           // tensor_dim1 hi | tile_dim0
  g1[4] = (int)((t1 & 0xFFFFu) | (t2 << 16));        // tile_dim1 | tile_dim2
  g1[5] = (int)(uint32_t)(s0 & 0xFFFFFFFFull);       // dim0_stride lo32
  g1[6] = (int)(uint32_t)(((s0 >> 32) & 0xFFFFull) | ((s1 & 0xFFFFull) << 16));
  g1[7] = (int)(uint32_t)((s1 >> 16) & 0xFFFFFFFFull);
  v4i g2; g2[0] = (int)td2; g2[1] = 0; g2[2] = 0; g2[3] = 0;  // tensor_dim2
  v4i g3 = {0, 0, 0, 0};
#if defined(__clang_major__) && (__clang_major__ >= 23)
  v8i gx = {0, 0, 0, 0, 0, 0, 0, 0};
  __builtin_amdgcn_tensor_load_to_lds(g0, g1, g2, g3, gx, 0);
#else
  __builtin_amdgcn_tensor_load_to_lds(g0, g1, g2, g3, 0);
#endif
}
#endif

// ---------------------------------------------------------------------------
__global__ void __launch_bounds__(256) cvt_f16_kernel(
    const float* __restrict__ in, _Float16* __restrict__ out)
{
  int i = blockIdx.x * 256 + threadIdx.x;
  out[i] = (_Float16)in[i];
}

// depthwise 3x3, stride 1, SAME; fp32 accumulate
template <typename TI, typename TO>
__global__ void __launch_bounds__(256) dw3x3_kernel(
    const TI* __restrict__ in, const float* __restrict__ w,
    const float* __restrict__ bias, TO* __restrict__ out, int C, int hasbias)
{
  int idx = blockIdx.x * 256 + threadIdx.x;
  int x  = idx & 63;
  int y  = (idx >> 6) & 63;
  int nc = idx >> 12;
  int ch = nc % C;
  const float* wp = w + ch * 9;
  const TI* ip = in + (size_t)nc * NPIX;
  float acc = hasbias ? bias[ch] : 0.0f;
  #pragma unroll
  for (int dy = -1; dy <= 1; ++dy) {
    int yy = y + dy;
    if (yy < 0 || yy > 63) continue;
    #pragma unroll
    for (int dx = -1; dx <= 1; ++dx) {
      int xx = x + dx;
      if (xx < 0 || xx > 63) continue;
      acc += (float)ip[yy * HWDIM + xx] * wp[(dy + 1) * 3 + (dx + 1)];
    }
  }
  out[idx] = (TO)acc;
}

// ---------------------------------------------------------------------------
// A fragment from row-major [row][TK] f16 LDS tile (ISA 7.12.2 16-bit A layout)
// ---------------------------------------------------------------------------
__device__ __forceinline__ v16h a_frag(const _Float16* Abase, int wrow, int lane) {
  int h = (lane >> 4) & 1;
  const _Float16* p = Abase + (wrow + (lane & 15)) * TK + 8 * h;
  v8h lo = *(const v8h*)(p);
  v8h hi = *(const v8h*)(p + 16);
  return __builtin_shufflevector(lo, hi, 0, 1, 2, 3, 4, 5, 6, 7,
                                 8, 9, 10, 11, 12, 13, 14, 15);
}

__device__ __forceinline__ float sigmoidf_(float x) { return 1.0f / (1.0f + __expf(-x)); }
__device__ __forceinline__ float relu6f_(float x)   { return fminf(fmaxf(x, 0.0f), 6.0f); }

// B staging (256 thr): col = tid&63, kq = tid>>6; lane fragment = 32 contig bytes.
__device__ __forceinline__ int b_sdst(int tid) {
  int scol = tid & 63, skq = tid >> 6;
  return (scol >> 4) * 512 + (((skq & 1) << 4) | (scol & 15)) * 16 + (skq >> 1) * 8;
}

// ---------------------------------------------------------------------------
// GEMM1: i = Wy.[xw;h] + b   — double-buffered software pipeline
// LDS halfs: A0@0(4096) A1@4096 B0@8192(2048) B1@10240  -> 24KB
// ---------------------------------------------------------------------------
__global__ void __launch_bounds__(256) gemm1_kernel(
    const _Float16* __restrict__ xw16, const float* __restrict__ hin,
    const _Float16* __restrict__ Wy16, const float* __restrict__ Wyb,
    _Float16* __restrict__ out16)
{
  __shared__ __attribute__((aligned(32))) _Float16 lds[12288];
  const int ASZ = TM1 * TK;            // 4096 halfs
  const int BOFF = 2 * ASZ;            // 8192
  const int BSZ = 4 * 512;             // 2048 halfs

  const int tid    = threadIdx.x;
  const int lane   = tid & 31;
  const int wave   = tid >> 5;
  const int pbase  = blockIdx.x * TN;
  const int cobase = blockIdx.y * TM1;
  const int batch  = blockIdx.z;
  const int wrow   = wave * 16;
  const int sdst   = b_sdst(tid);
  const int pcs    = pbase + (tid & 63);
  const int skq    = tid >> 6;

  const _Float16* xwb = xw16 + (size_t)batch * CIN_ * NPIX;
  const float*    hb  = hin  + (size_t)batch * CH_  * NPIX;

  const v8f vz = {0.f, 0.f, 0.f, 0.f, 0.f, 0.f, 0.f, 0.f};
  v8f acc[4] = {vz, vz, vz, vz};

  // ---- B-source loader (phase select at the 320 boundary; 320 % 32 == 0)
  auto load_bvec = [&](int k0) -> v8h {
    v8h bv;
    if (k0 < CIN_) {
      const _Float16* s = xwb + (size_t)(k0 + skq * 8) * NPIX + pcs;
      #pragma unroll
      for (int j = 0; j < 8; ++j) bv[j] = s[(size_t)j * NPIX];
    } else {
      const float* s = hb + (size_t)(k0 - CIN_ + skq * 8) * NPIX + pcs;
      #pragma unroll
      for (int j = 0; j < 8; ++j) bv[j] = (_Float16)s[(size_t)j * NPIX];
    }
    return bv;
  };
  auto issue_A = [&](int buf, int k0) {
#if HAVE_TDM
    if (tid < 32)
      tdm_load_f16((uint32_t)(buf * ASZ * 2),
                   (uint64_t)(uintptr_t)Wy16 + 2ull * ((size_t)cobase * KCAT + k0),
                   KCAT, CH_, 0, KCAT, 0, TK, TM1, 0);
#else
    _Float16* dst = lds + buf * ASZ;
    #pragma unroll
    for (int j = 0; j < ASZ / 256; ++j) {
      int lin = tid * (ASZ / 256) + j;
      int row = lin >> 5, kk = lin & 31;
      dst[row * TK + kk] = Wy16[(size_t)(cobase + row) * KCAT + k0 + kk];
    }
#endif
  };

  const int nk = KCAT / TK;            // 26
  // prologue: tile 0 into buffer 0
  {
    v8h bv = load_bvec(0);
    *(v8h*)(lds + BOFF + sdst) = bv;
    issue_A(0, 0);
  }
  #pragma unroll 1
  for (int k = 0; k < nk; ++k) {
    const int cur = k & 1;
#if HAVE_TDM
    if (tid < 32) __builtin_amdgcn_s_wait_tensorcnt(0);   // A(k) landed
#endif
    __syncthreads();                                      // A(k), B(k) visible
    const bool more = (k + 1) < nk;
    if (more) issue_A(cur ^ 1, (k + 1) * TK);             // DMA overlaps compute
    v8h bnext = {};
    if (more) bnext = load_bvec((k + 1) * TK);            // global loads in flight
    // ---- WMMA on tile k
    v16h a = a_frag(lds + cur * ASZ, wrow, lane);
    const v16h* bfp = (const v16h*)(lds + BOFF + cur * BSZ);
    #pragma unroll
    for (int t = 0; t < 4; ++t) {
      v16h b = bfp[t * 32 + lane];
      acc[t] = __builtin_amdgcn_wmma_f32_16x16x32_f16(
          false, a, false, b, (short)0, acc[t], false, false);
    }
    if (more) *(v8h*)(lds + BOFF + (cur ^ 1) * BSZ + sdst) = bnext;
  }
  // epilogue: +bias, f16 store
  const int hh  = (lane >> 4) & 1;
  const int col = lane & 15;
  _Float16* op = out16 + (size_t)batch * CH_ * NPIX;
  #pragma unroll
  for (int t = 0; t < 4; ++t)
    #pragma unroll
    for (int r = 0; r < 8; ++r) {
      int row = cobase + wrow + 8 * hh + r;
      op[(size_t)row * NPIX + pbase + t * 16 + col] = (_Float16)(acc[t][r] + Wyb[row]);
    }
}

// ---------------------------------------------------------------------------
// Fused 4-gate GEMM + LSTM cell update — double-buffered, TM=64, waves 4(M)x2(N).
// One 3-D TDM per K step loads all four 64x32 gate tiles (tile_dim2=4).
// LDS halfs: A0@0(8192) A1@8192 B0@16384(2048) B1@18432 -> 40KB
// ---------------------------------------------------------------------------
__global__ void __launch_bounds__(256) gates_kernel(
    const _Float16* __restrict__ bact16, const float* __restrict__ cprev,
    const _Float16* __restrict__ Wg16,   // [4][CH][CH] f16 contiguous
    float* __restrict__ out_ch, float* __restrict__ out_cc)
{
  __shared__ __attribute__((aligned(32))) _Float16 lds[20480];
  const int AGATE = TMG * TK;          // 2048 halfs per gate tile
  const int ASZ = 4 * AGATE;           // 8192 halfs per buffer
  const int BOFF = 2 * ASZ;            // 16384
  const int BSZ = 4 * 512;             // 2048

  const int tid    = threadIdx.x;
  const int lane   = tid & 31;
  const int wave   = tid >> 5;
  const int wm     = wave & 3;         // M group (16 rows)
  const int wc     = wave >> 2;        // N half (32 cols)
  const int pbase  = blockIdx.x * TN;
  const int cobase = blockIdx.y * TMG;
  const int batch  = blockIdx.z;
  const int wrow   = wm * 16;
  const int sdst   = b_sdst(tid);
  const int pcs    = pbase + (tid & 63);
  const int skq    = tid >> 6;

  const _Float16* bb = bact16 + (size_t)batch * CH_ * NPIX;

  const v8f vz = {0.f, 0.f, 0.f, 0.f, 0.f, 0.f, 0.f, 0.f};
  v8f accI[2] = {vz, vz};
  v8f accF[2] = {vz, vz};
  v8f accG[2] = {vz, vz};
  v8f accO[2] = {vz, vz};

  auto load_bvec = [&](int k0) -> v8h {
    const _Float16* s = bb + (size_t)(k0 + skq * 8) * NPIX + pcs;
    v8h bv;
    #pragma unroll
    for (int j = 0; j < 8; ++j) bv[j] = s[(size_t)j * NPIX];
    return bv;
  };
  auto issue_A = [&](int buf, int k0) {
#if HAVE_TDM
    if (tid < 32)
      tdm_load_f16((uint32_t)(buf * ASZ * 2),
                   (uint64_t)(uintptr_t)Wg16 + 2ull * ((size_t)cobase * CH_ + k0),
                   CH_, CH_, 4,
                   CH_, (uint64_t)CH_ * CH_,
                   TK, TMG, 4);
#else
    _Float16* dst = lds + buf * ASZ;
    #pragma unroll
    for (int g = 0; g < 4; ++g)
      #pragma unroll
      for (int j = 0; j < AGATE / 256; ++j) {
        int lin = tid * (AGATE / 256) + j;
        int row = lin >> 5, kk = lin & 31;
        dst[g * AGATE + row * TK + kk] =
            Wg16[(size_t)g * CH_ * CH_ + (size_t)(cobase + row) * CH_ + k0 + kk];
      }
#endif
  };

  const int nk = CH_ / TK;             // 16
  {
    v8h bv = load_bvec(0);
    *(v8h*)(lds + BOFF + sdst) = bv;
    issue_A(0, 0);
  }
  #pragma unroll 1
  for (int k = 0; k < nk; ++k) {
    const int cur = k & 1;
#if HAVE_TDM
    if (tid < 32) __builtin_amdgcn_s_wait_tensorcnt(0);
#endif
    __syncthreads();
    const bool more = (k + 1) < nk;
    if (more) issue_A(cur ^ 1, (k + 1) * TK);
    v8h bnext = {};
    if (more) bnext = load_bvec((k + 1) * TK);

    const _Float16* ab = lds + cur * ASZ;
    v16h aI = a_frag(ab + 0 * AGATE, wrow, lane);
    v16h aF = a_frag(ab + 1 * AGATE, wrow, lane);
    v16h aG = a_frag(ab + 2 * AGATE, wrow, lane);
    v16h aO = a_frag(ab + 3 * AGATE, wrow, lane);
    const v16h* bfp = (const v16h*)(lds + BOFF + cur * BSZ);
    #pragma unroll
    for (int t = 0; t < 2; ++t) {
      v16h b = bfp[(wc * 2 + t) * 32 + lane];
      accI[t] = __builtin_amdgcn_wmma_f32_16x16x32_f16(false, aI, false, b, (short)0, accI[t], false, false);
      accF[t] = __builtin_amdgcn_wmma_f32_16x16x32_f16(false, aF, false, b, (short)0, accF[t], false, false);
      accG[t] = __builtin_amdgcn_wmma_f32_16x16x32_f16(false, aG, false, b, (short)0, accG[t], false, false);
      accO[t] = __builtin_amdgcn_wmma_f32_16x16x32_f16(false, aO, false, b, (short)0, accO[t], false, false);
    }
    if (more) *(v8h*)(lds + BOFF + (cur ^ 1) * BSZ + sdst) = bnext;
  }
  // fused LSTM epilogue
  const int hh  = (lane >> 4) & 1;
  const int col = lane & 15;
  const size_t base = (size_t)batch * CH_ * NPIX;
  #pragma unroll
  for (int t = 0; t < 2; ++t)
    #pragma unroll
    for (int r = 0; r < 8; ++r) {
      int row = cobase + wrow + 8 * hh + r;
      size_t off = base + (size_t)row * NPIX + pbase + (wc * 2 + t) * 16 + col;
      float cv  = cprev[off];
      float ci  = sigmoidf_(accI[t][r]);
      float cf  = sigmoidf_(accF[t][r]);
      float g6  = relu6f_(accG[t][r]);
      float cog = sigmoidf_(accO[t][r]);
      float cc  = cf * cv + ci * g6;
      out_cc[off] = cc;
      out_ch[off] = cog * relu6f_(cc);
    }
}

// ---------------------------------------------------------------------------
extern "C" void kernel_launch(void* const* d_in, const int* in_sizes, int n_in,
                              void* d_out, int out_size, void* d_ws, size_t ws_size,
                              hipStream_t stream) {
  const float* x     = (const float*)d_in[0];
  const float* h     = (const float*)d_in[1];
  const float* c     = (const float*)d_in[2];
  const float* W_dw  = (const float*)d_in[3];
  const float* W_dwb = (const float*)d_in[4];
  const float* Wy    = (const float*)d_in[5];
  const float* Wy_b  = (const float*)d_in[6];
  const float* Wi    = (const float*)d_in[7];
  const float* Wgates[4] = {(const float*)d_in[8], (const float*)d_in[9],
                            (const float*)d_in[10], (const float*)d_in[11]};

  _Float16* wsh  = (_Float16*)d_ws;
  _Float16* xw16 = wsh;                                  // 8*320*4096
  _Float16* i16  = xw16 + (size_t)BATCH * CIN_ * NPIX;   // 8*512*4096
  _Float16* b16  = i16  + (size_t)BATCH * CH_  * NPIX;   // 8*512*4096
  _Float16* Wy16 = b16  + (size_t)BATCH * CH_  * NPIX;   // 512*832
  _Float16* Wg16 = Wy16 + (size_t)CH_ * KCAT;            // 4*512*512

  float* out_ch = (float*)d_out;
  float* out_cc = out_ch + (size_t)BATCH * CH_ * NPIX;

  // 0) weights -> f16 (tiny; enables raw-byte TDM tile loads)
  cvt_f16_kernel<<<(CH_ * KCAT) / 256, 256, 0, stream>>>(Wy, Wy16);
  for (int g = 0; g < 4; ++g)
    cvt_f16_kernel<<<(CH_ * CH_) / 256, 256, 0, stream>>>(
        Wgates[g], Wg16 + (size_t)g * CH_ * CH_);

  // 1) xw = dw3x3(x) + bias   (fp32 in, f16 out)
  dw3x3_kernel<float, _Float16><<<(BATCH * CIN_ * NPIX) / 256, 256, 0, stream>>>(
      x, W_dw, W_dwb, xw16, CIN_, 1);

  // 2) i = Wy.[xw;h] + b      (WMMA, pipelined TDM weight tiles)
  dim3 g1(NPIX / TN, CH_ / TM1, BATCH);
  gemm1_kernel<<<g1, 256, 0, stream>>>(xw16, h, Wy16, Wy_b, i16);

  // 3) b = dw3x3(i)           (f16 in, f16 out)
  dw3x3_kernel<_Float16, _Float16><<<(BATCH * CH_ * NPIX) / 256, 256, 0, stream>>>(
      i16, Wi, (const float*)nullptr, b16, CH_, 0);

  // 4) fused 4-gate GEMM + LSTM cell update -> (ch, cc)
  dim3 g2(NPIX / TN, CH_ / TMG, BATCH);
  gates_kernel<<<g2, 256, 0, stream>>>(b16, c, Wg16, out_ch, out_cc);
}
